// TransformerLMHead_4243427689116
// MI455X (gfx1250) — matrix operand order repaired and checked by reading, not compiled, not run on hardware
//
#include <hip/hip_runtime.h>
#include <cstdint>

// ---------------------------------------------------------------------------
// CDNA5 WMMA (wave32):  D(16x16 f32) = A(16x32 bf16) x B(32x16 bf16) + C
// ---------------------------------------------------------------------------
typedef __bf16 bf16;
typedef __attribute__((ext_vector_type(16))) __bf16 bf16x16;
typedef __attribute__((ext_vector_type(8)))  float  floatx8;

__device__ __forceinline__ bf16 f2bf(float f) {
  unsigned u = __builtin_bit_cast(unsigned, f);
  unsigned r = (u + 0x7FFFu + ((u >> 16) & 1u)) >> 16;   // round-to-nearest-even
  unsigned short h = (unsigned short)r;
  return __builtin_bit_cast(bf16, h);
}

// A-operand: lanes 0-15 hold K 0..15, lanes 16-31 K 16..31 (contiguous)
__device__ __forceinline__ bf16x16 load_frag2(const bf16* p0, const bf16* p1) {
  union { bf16x16 v; uint4 q[2]; } u;
  u.q[0] = *(const uint4*)p0;
  u.q[1] = *(const uint4*)p1;
  return u.v;
}
// B-operand: lane = column N (lane&15); K = (lane>>4)*16 + 0..15 contiguous
__device__ __forceinline__ bf16x16 load_frag1(const bf16* p) {
  union { bf16x16 v; uint4 q[2]; } u;
  u.q[0] = ((const uint4*)p)[0];
  u.q[1] = ((const uint4*)p)[1];
  return u.v;
}

__device__ __forceinline__ floatx8 wmma_bf16(bf16x16 a, bf16x16 b, floatx8 c) {
  return __builtin_amdgcn_wmma_f32_16x16x32_bf16(false, a, false, b,
                                                 (short)0, c, false, false);
}

// Async DMA: global -> LDS, 16 bytes/lane, tracked by ASYNCcnt (CDNA5 §15.18).
__device__ __forceinline__ void async_cp16(unsigned lds_off, const void* gptr) {
  uint64_t ga = (uint64_t)(size_t)gptr;
  asm volatile("global_load_async_to_lds_b128 %0, %1, off"
               :: "v"(lds_off), "v"(ga) : "memory");
}
__device__ __forceinline__ void wait_async0() {
  asm volatile("s_wait_asynccnt 0x0" ::: "memory");
}
__device__ __forceinline__ unsigned lds_addr(const void* p) {
  return (unsigned)(size_t)p;     // flat LDS addr: low 32 bits = LDS offset
}

// ---------------------------------------------------------------------------
// Weight prep: W[K,N] f32  ->  Wt[N,K] bf16  (tiled transpose through LDS)
// grid (N/32, K/32), 256 threads (32x8)
// ---------------------------------------------------------------------------
__global__ __launch_bounds__(256) void wprep_k(const float* __restrict__ W,
                                               bf16* __restrict__ Wt,
                                               int K, int N) {
  __shared__ float t[32][33];
  const int n0 = blockIdx.x * 32, k0 = blockIdx.y * 32;
  const int lx = threadIdx.x & 31, ly = threadIdx.x >> 5;
#pragma unroll
  for (int i = 0; i < 32; i += 8)
    t[ly + i][lx] = W[(size_t)(k0 + ly + i) * N + n0 + lx];
  __syncthreads();
#pragma unroll
  for (int i = 0; i < 32; i += 8)
    Wt[(size_t)(n0 + ly + i) * K + k0 + lx] = f2bf(t[lx][ly + i]);
}

// ---------------------------------------------------------------------------
// Embedding: x[b,s,:] = emb[tok[b,s],:] + pos[s,:]   (f32 + bf16 mirror)
// ---------------------------------------------------------------------------
__global__ __launch_bounds__(256) void embed_k(const int* __restrict__ tok,
                                               const float* __restrict__ emb,
                                               const float* __restrict__ pos,
                                               float* __restrict__ x,
                                               bf16* __restrict__ xb) {
  int row = blockIdx.x;
  int s   = row & 2047;
  int t   = tok[row];
  int d   = threadIdx.x * 4;
#pragma unroll
  for (int i = 0; i < 4; ++i) {
    float v = emb[(size_t)t * 1024 + d + i] + pos[(size_t)s * 1024 + d + i];
    x [(size_t)row * 1024 + d + i] = v;
    xb[(size_t)row * 1024 + d + i] = f2bf(v);
  }
}

// ---------------------------------------------------------------------------
// bf16 WMMA GEMM:  C[M,N] = A[M,K](bf16) * Wt[N,K](bf16, pre-transposed) + bias
// 128x128 block tile, BK=32, 256 thr = 8 waves (2x4), wave tile 64x32.
// Double-buffered LDS, tiles staged with async DMA overlapped with WMMA.
// MODE 0: bf16 | 1: gelu->bf16 | 2: +resid -> f32+bf16 | 3: f32
// MODE 4: bf16 transposed to [b][col][s] (for attention V)
// ---------------------------------------------------------------------------
template <int MODE>
__global__ __launch_bounds__(256) void gemm_k(const bf16* __restrict__ A,
                                              const bf16* __restrict__ Wt,
                                              const float* __restrict__ bias,
                                              const float* __restrict__ resid,
                                              float* __restrict__ Cf,
                                              bf16* __restrict__ Cb,
                                              int M, int N, int K) {
  __shared__ __align__(16) bf16 As[2][128 * 32];   // [row][k]  2 x 8 KB
  __shared__ __align__(16) bf16 Ws[2][128 * 32];   // [n][k]    2 x 8 KB

  const int tid  = threadIdx.x;
  const int lane = tid & 31, w = tid >> 5;
  const int hi   = lane >> 4, l16 = lane & 15;
  const int wm   = w >> 2, wn = w & 3;
  const int rowBase = blockIdx.y * 128;
  const int colBase = blockIdx.x * 128;

  const int srow = tid >> 1, skc = (tid & 1) * 16;   // staging: 32B per thread

  floatx8 acc[4][2] = {};

  auto issue = [&](int buf, int k0) {
    const bf16* ga = A  + (size_t)(rowBase + srow) * K + k0 + skc;
    const bf16* gw = Wt + (size_t)(colBase + srow) * K + k0 + skc;
    unsigned la = lds_addr(&As[buf][srow * 32 + skc]);
    unsigned lw = lds_addr(&Ws[buf][srow * 32 + skc]);
    async_cp16(la,      ga);
    async_cp16(la + 16, ga + 8);
    async_cp16(lw,      gw);
    async_cp16(lw + 16, gw + 8);
  };

  const int nk = K >> 5;
  issue(0, 0);
  for (int ki = 0; ki < nk; ++ki) {
    const int cur = ki & 1;
    wait_async0();
    __syncthreads();
    if (ki + 1 < nk) issue(cur ^ 1, (ki + 1) << 5);   // DMA overlaps WMMA

    bf16x16 bfr[2];
#pragma unroll
    for (int ni = 0; ni < 2; ++ni) {
      int n = wn * 32 + ni * 16 + l16;
      bfr[ni] = load_frag1(&Ws[cur][n * 32 + hi * 16]);
    }
#pragma unroll
    for (int mi = 0; mi < 4; ++mi) {
      int r = wm * 64 + mi * 16 + l16;
      bf16x16 afr = load_frag2(&As[cur][r * 32 + hi * 16],
                               &As[cur][r * 32 + hi * 16 + 8]);
#pragma unroll
      for (int ni = 0; ni < 2; ++ni)
        acc[mi][ni] = wmma_bf16(afr, bfr[ni], acc[mi][ni]);
    }
  }

  // epilogue (C layout: lanes 0-15 rows 0-7, lanes 16-31 rows 8-15; col=lane&15)
#pragma unroll
  for (int ni = 0; ni < 2; ++ni) {
    int col = colBase + wn * 32 + ni * 16 + l16;
    float bv = bias ? bias[col] : 0.f;
#pragma unroll
    for (int mi = 0; mi < 4; ++mi) {
#pragma unroll
      for (int r = 0; r < 8; ++r) {
        int row = rowBase + wm * 64 + mi * 16 + hi * 8 + r;
        float v = acc[mi][ni][r] + bv;
        size_t idx = (size_t)row * N + col;
        if constexpr (MODE == 0) {
          Cb[idx] = f2bf(v);
        } else if constexpr (MODE == 1) {
          float g = 0.5f * v * (1.f + erff(v * 0.70710678118654752f));
          Cb[idx] = f2bf(g);
        } else if constexpr (MODE == 2) {
          v += resid[idx];
          Cf[idx] = v;
          Cb[idx] = f2bf(v);
        } else if constexpr (MODE == 3) {
          Cf[idx] = v;
        } else {  // MODE 4: [b][col][s] transposed (V for attention)
          size_t tix = ((size_t)(row >> 11) * 1024 + col) * 2048 + (row & 2047);
          Cb[tix] = f2bf(v);
        }
      }
    }
  }
}

// ---------------------------------------------------------------------------
// Flash attention (causal): 4 waves x 16 queries, 32-key blocks, online
// softmax; K/V staged by async DMA (double-buffered); WMMA for QK^T and P*V.
// q/k: [(b*S+s), h*64+d] bf16.  v: pre-transposed [b][h*64+dv][s] bf16.
// ---------------------------------------------------------------------------
__global__ __launch_bounds__(128) void attn_k(const bf16* __restrict__ Q,
                                              const bf16* __restrict__ Kb,
                                              const bf16* __restrict__ Vt,
                                              bf16* __restrict__ O) {
  __shared__ __align__(16) bf16 Ks[2][32 * 64];    // [krel][d]   2 x 4 KB
  __shared__ __align__(16) bf16 Vs[2][64 * 32];    // [dv][krel]  2 x 4 KB
  __shared__ __align__(16) bf16 Ps[4][16 * 32];    // per-wave P  4 KB

  const int tid  = threadIdx.x;
  const int lane = tid & 31, w = tid >> 5;
  const int hi   = lane >> 4, l16 = lane & 15;
  const int bh   = blockIdx.x;
  const int b    = bh >> 4, h = bh & 15;
  const int qblk = blockIdx.y;
  const int q0   = qblk * 64 + w * 16;
  const size_t RS    = 1024;
  const size_t base  = (size_t)b * 2048 * RS + (size_t)h * 64;     // q/k/o
  const size_t vbase = ((size_t)b * 1024 + h * 64) * 2048;         // v (T)

  bf16x16 aq[2];
  {
    const bf16* qp = Q + base + (size_t)(q0 + l16) * RS;
#pragma unroll
    for (int dh = 0; dh < 2; ++dh)
      aq[dh] = load_frag2(qp + dh * 32 + hi * 16, qp + dh * 32 + hi * 16 + 8);
  }

  floatx8 acc[4] = {};
  float mrow[8], lrow[8];
#pragma unroll
  for (int r = 0; r < 8; ++r) { mrow[r] = -1e30f; lrow[r] = 0.f; }

  const int krow = tid >> 2, kdc = (tid & 3) * 16;   // K stage: 32 x 64
  const int vdv  = tid >> 1, vkc = (tid & 1) * 16;   // V stage: 64 x 32

  auto issue = [&](int buf, int kb) {
    const bf16* gk = Kb + base + (size_t)(kb + krow) * RS + kdc;
    unsigned lk = lds_addr(&Ks[buf][krow * 64 + kdc]);
    async_cp16(lk,      gk);
    async_cp16(lk + 16, gk + 8);
    const bf16* gv = Vt + vbase + (size_t)vdv * 2048 + kb + vkc;
    unsigned lv = lds_addr(&Vs[buf][vdv * 32 + vkc]);
    async_cp16(lv,      gv);
    async_cp16(lv + 16, gv + 8);
  };

  const int kend = qblk * 64 + 64;
  issue(0, 0);
  for (int kb = 0; kb < kend; kb += 32) {
    const int cur = (kb >> 5) & 1;
    wait_async0();
    __syncthreads();
    if (kb + 32 < kend) issue(cur ^ 1, kb + 32);

    // scores: two 16(q) x 16(key) tiles, DK=64 => 2 WMMAs each
    floatx8 s[2];
#pragma unroll
    for (int kt = 0; kt < 2; ++kt) {
      int key = kt * 16 + l16;
      bf16x16 b0 = load_frag1(&Ks[cur][key * 64 + hi * 16]);
      bf16x16 b1 = load_frag1(&Ks[cur][key * 64 + 32 + hi * 16]);
      floatx8 c = {};
      c = wmma_bf16(aq[0], b0, c);
      c = wmma_bf16(aq[1], b1, c);
      s[kt] = c;
    }

    // online softmax (rows split by half-wave; 16-lane xor reductions)
    float p0[8], p1[8];
#pragma unroll
    for (int r = 0; r < 8; ++r) {
      int qrow = q0 + hi * 8 + r;
      float s0 = s[0][r] * 0.125f;
      float s1 = s[1][r] * 0.125f;
      if (kb + l16 > qrow)      s0 = -1e30f;
      if (kb + 16 + l16 > qrow) s1 = -1e30f;
      float t = fmaxf(s0, s1);
#pragma unroll
      for (int m = 1; m < 16; m <<= 1) t = fmaxf(t, __shfl_xor(t, m, 32));
      float mn = fmaxf(mrow[r], t);
      float alpha = __expf(mrow[r] - mn);
      float e0 = __expf(s0 - mn);
      float e1 = __expf(s1 - mn);
      float rs = e0 + e1;
#pragma unroll
      for (int m = 1; m < 16; m <<= 1) rs += __shfl_xor(rs, m, 32);
      lrow[r] = lrow[r] * alpha + rs;
      mrow[r] = mn;
#pragma unroll
      for (int nt = 0; nt < 4; ++nt) acc[nt][r] *= alpha;
      p0[r] = e0; p1[r] = e1;
    }

    // C-layout P -> per-wave LDS -> A-layout (DS in-order per wave + wait)
    bf16* ps = &Ps[w][0];
#pragma unroll
    for (int r = 0; r < 8; ++r) {
      ps[(hi * 8 + r) * 32 + l16]      = f2bf(p0[r]);
      ps[(hi * 8 + r) * 32 + 16 + l16] = f2bf(p1[r]);
    }
    asm volatile("s_wait_dscnt 0x0" ::: "memory");
    bf16x16 ap = load_frag2(&ps[l16 * 32 + hi * 16], &ps[l16 * 32 + hi * 16 + 8]);

    // P(16x32) x V(32x16) for the four dv tiles
#pragma unroll
    for (int nt = 0; nt < 4; ++nt) {
      bf16x16 bv = load_frag1(&Vs[cur][(nt * 16 + l16) * 32 + hi * 16]);
      acc[nt] = wmma_bf16(ap, bv, acc[nt]);
    }
  }

#pragma unroll
  for (int nt = 0; nt < 4; ++nt)
#pragma unroll
    for (int r = 0; r < 8; ++r) {
      int qrow = q0 + hi * 8 + r;
      float v = acc[nt][r] / lrow[r];
      O[base + (size_t)qrow * RS + nt * 16 + l16] = f2bf(v);
    }
}

// ---------------------------------------------------------------------------
// Host orchestration
// ---------------------------------------------------------------------------
extern "C" void kernel_launch(void* const* d_in, const int* in_sizes, int n_in,
                              void* d_out, int out_size, void* d_ws, size_t ws_size,
                              hipStream_t stream) {
  const int*   tokens = (const int*)  d_in[0];
  const float* emb    = (const float*)d_in[1];
  const float* pos    = (const float*)d_in[2];
  const float* Wq     = (const float*)d_in[3];
  const float* bq     = (const float*)d_in[4];
  const float* Wk     = (const float*)d_in[5];
  const float* bk     = (const float*)d_in[6];
  const float* Wv     = (const float*)d_in[7];
  const float* bv     = (const float*)d_in[8];
  const float* Wo     = (const float*)d_in[9];
  const float* W1     = (const float*)d_in[10];
  const float* b1     = (const float*)d_in[11];
  const float* W2     = (const float*)d_in[12];
  const float* b2     = (const float*)d_in[13];
  const float* Wr     = (const float*)d_in[14];
  const float* br     = (const float*)d_in[15];

  const int M = 4096;                    // B*S
  const int D = 1024, DH = 4096, Vv = 256, L = 16;

  char* ws = (char*)d_ws;
  const size_t MB = 1u << 20;
  float* x0   = (float*)(ws + 0 * MB);
  float* x1   = (float*)(ws + 16 * MB);
  float* hbuf = (float*)(ws + 32 * MB);
  bf16*  x0b  = (bf16*)(ws + 48 * MB);
  bf16*  x1b  = (bf16*)(ws + 56 * MB);
  bf16*  hb   = (bf16*)(ws + 64 * MB);
  bf16*  qb   = (bf16*)(ws + 72 * MB);
  bf16*  kbf  = (bf16*)(ws + 80 * MB);
  bf16*  vt   = (bf16*)(ws + 88 * MB);   // V transposed [b][h*64+dv][s]
  bf16*  ob   = (bf16*)(ws + 96 * MB);
  bf16*  mb   = (bf16*)(ws + 104 * MB);  // 32 MB
  bf16*  wqT  = (bf16*)(ws + 136 * MB);  // per-layer bf16 weights [N][K]
  bf16*  wkT  = (bf16*)(ws + 138 * MB);
  bf16*  wvT  = (bf16*)(ws + 140 * MB);
  bf16*  woT  = (bf16*)(ws + 142 * MB);
  bf16*  w1T  = (bf16*)(ws + 144 * MB);  // 8 MB  [4096][1024]
  bf16*  w2T  = (bf16*)(ws + 152 * MB);  // 8 MB  [1024][4096]
  bf16*  wrT  = (bf16*)(ws + 160 * MB);  // 0.5MB [256][1024]

  embed_k<<<M, 256, 0, stream>>>(tokens, emb, pos, x0, x0b);

  float* x = x0;  bf16* xbf = x0b;
  float* xn = x1; bf16* xnb = x1b;

  dim3 g8(8, 32), g32(32, 32), g2(2, 32);      // (N/128, M/128)
  dim3 tDD(32, 32), tDH(128, 32), tHD(32, 128), tDV(8, 32);

  for (int l = 0; l < L; ++l) {
    const float* wq = Wq + (size_t)l * D * 1024;  const float* bq_ = bq + (size_t)l * 1024;
    const float* wk = Wk + (size_t)l * D * 1024;  const float* bk_ = bk + (size_t)l * 1024;
    const float* wv = Wv + (size_t)l * D * 1024;  const float* bv_ = bv + (size_t)l * 1024;
    const float* wo = Wo + (size_t)l * 1024 * D;
    const float* w1 = W1 + (size_t)l * D * DH;    const float* b1_ = b1 + (size_t)l * DH;
    const float* w2 = W2 + (size_t)l * DH * D;    const float* b2_ = b2 + (size_t)l * D;

    // one-time per-layer weight transpose+bf16 (L2-resident afterwards)
    wprep_k<<<tDD, 256, 0, stream>>>(wq, wqT, D, 1024);
    wprep_k<<<tDD, 256, 0, stream>>>(wk, wkT, D, 1024);
    wprep_k<<<tDD, 256, 0, stream>>>(wv, wvT, D, 1024);
    wprep_k<<<tDD, 256, 0, stream>>>(wo, woT, 1024, D);
    wprep_k<<<tDH, 256, 0, stream>>>(w1, w1T, D, DH);
    wprep_k<<<tHD, 256, 0, stream>>>(w2, w2T, DH, D);

    gemm_k<0><<<g8, 256, 0, stream>>>(xbf, wqT, bq_, nullptr, nullptr, qb, M, 1024, D);
    gemm_k<0><<<g8, 256, 0, stream>>>(xbf, wkT, bk_, nullptr, nullptr, kbf, M, 1024, D);
    gemm_k<4><<<g8, 256, 0, stream>>>(xbf, wvT, bv_, nullptr, nullptr, vt, M, 1024, D);

    attn_k<<<dim3(32, 32), 128, 0, stream>>>(qb, kbf, vt, ob);

    // h = x + o @ Wo (no bias)
    gemm_k<2><<<g8, 256, 0, stream>>>(ob, woT, nullptr, x, hbuf, hb, M, 1024, 1024);
    // hidden = gelu(h @ W1 + b1)
    gemm_k<1><<<g32, 256, 0, stream>>>(hb, w1T, b1_, nullptr, nullptr, mb, M, DH, D);
    // x_new = x + hidden @ W2 + b2   (residual from x, per reference)
    gemm_k<2><<<g8, 256, 0, stream>>>(mb, w2T, b2_, x, xn, xnb, M, 1024, DH);

    float* tf = x; x = xn; xn = tf;
    bf16* tb = xbf; xbf = xnb; xnb = tb;
  }

  wprep_k<<<tDV, 256, 0, stream>>>(Wr, wrT, D, Vv);
  gemm_k<3><<<g2, 256, 0, stream>>>(xbf, wrT, br, nullptr, (float*)d_out, nullptr, M, Vv, D);
}
